// LoRALinear_46918222742185
// MI455X (gfx1250) — compile-verified, matrix operand order checked by reading, not run on hardware
//
#include <hip/hip_runtime.h>

// ---- types ----
typedef __attribute__((ext_vector_type(16))) _Float16 v16h;
typedef __attribute__((ext_vector_type(8)))  _Float16 v8h;
typedef __attribute__((ext_vector_type(8)))  float    v8f;
typedef __attribute__((ext_vector_type(4)))  float    v4f;
typedef __attribute__((ext_vector_type(4)))  int      v4i;

#define IN_DIM   4096
#define OUT_DIM  4096
#define M_TOK    8192            // 4 * 2048
#define RANK     16
#define NF_BLOCK 64
#define LDA      40              // LDS row pitch in halves (80B: conflict-free b128)

__constant__ float NF4_LUT[16] = {
    -1.0f, -0.6961928009986877f, -0.5250730514526367f, -0.39491748809814453f,
    -0.28444138169288635f, -0.18477343022823334f, -0.09105003625154495f, 0.0f,
    0.07958029955625534f, 0.16093020141124725f, 0.24611230194568634f,
    0.33791524171829224f, 0.44070982933044434f, 0.5626170039176941f,
    0.7229568362236023f, 1.0f};

// Raw LDS byte offset of a shared-memory pointer (generic -> addrspace(3)).
__device__ __forceinline__ unsigned lds_offset(const void* p) {
    return (unsigned)(unsigned long long)(__attribute__((address_space(3))) const void*)p;
}

// Async Global->LDS 16-byte copy (per lane), tracked by ASYNCcnt.
__device__ __forceinline__ void async_copy_b128(unsigned ldsOff, const void* gptr) {
    asm volatile("global_load_async_to_lds_b128 %0, %1, off"
                 :: "v"(ldsOff), "v"(gptr) : "memory");
}

__device__ __forceinline__ void wait_asynccnt0() {
    asm volatile("s_wait_asynccnt 0x0" ::: "memory");
}

// ---------------------------------------------------------------------------
// Kernel 1: NF4 dequant -> f16 weights, row-major [OUT_DIM][IN_DIM]
// 8 codes per thread; all 8 share one absmax scale (8 | 64).
// ---------------------------------------------------------------------------
__global__ __launch_bounds__(256)
void dequant_nf4_kernel(const int* __restrict__ wq,
                        const float* __restrict__ scales,
                        _Float16* __restrict__ wh) {
    const int t    = blockIdx.x * 256 + threadIdx.x;   // 2,097,152 threads
    const int base = t * 8;
    const int o    = base >> 12;                       // / IN_DIM
    const int k    = base & (IN_DIM - 1);
    const float s  = scales[o * (IN_DIM / NF_BLOCK) + (k >> 6)];

    v4i c0 = *(const v4i*)(wq + base);
    v4i c1 = *(const v4i*)(wq + base + 4);

    v8h out = {
        (_Float16)(NF4_LUT[c0.x & 15] * s), (_Float16)(NF4_LUT[c0.y & 15] * s),
        (_Float16)(NF4_LUT[c0.z & 15] * s), (_Float16)(NF4_LUT[c0.w & 15] * s),
        (_Float16)(NF4_LUT[c1.x & 15] * s), (_Float16)(NF4_LUT[c1.y & 15] * s),
        (_Float16)(NF4_LUT[c1.z & 15] * s), (_Float16)(NF4_LUT[c1.w & 15] * s)};
    *(v8h*)(wh + base) = out;
}

// ---------------------------------------------------------------------------
// Kernel 2: bh[n][0:16] = 2.0 * lora_b[n][0:16] (f16), bh[n][16:32] = 0
// Folds ALPHA/RANK and zero-pads so LoRA becomes one extra GEMM K-step.
// ---------------------------------------------------------------------------
__global__ __launch_bounds__(256)
void pack_lorab_kernel(const float* __restrict__ lb, _Float16* __restrict__ bh) {
    const int n = blockIdx.x * 256 + threadIdx.x;      // 4096 threads
    const float* s = lb + n * RANK;
    v8h o0 = {(_Float16)(2.f*s[0]), (_Float16)(2.f*s[1]), (_Float16)(2.f*s[2]),
              (_Float16)(2.f*s[3]), (_Float16)(2.f*s[4]), (_Float16)(2.f*s[5]),
              (_Float16)(2.f*s[6]), (_Float16)(2.f*s[7])};
    v8h o1 = {(_Float16)(2.f*s[8]),  (_Float16)(2.f*s[9]),  (_Float16)(2.f*s[10]),
              (_Float16)(2.f*s[11]), (_Float16)(2.f*s[12]), (_Float16)(2.f*s[13]),
              (_Float16)(2.f*s[14]), (_Float16)(2.f*s[15])};
    v8h z = {};
    v8h* dst = (v8h*)(bh + n * 32);
    dst[0] = o0; dst[1] = o1; dst[2] = z; dst[3] = z;
}

// ---------------------------------------------------------------------------
// Kernel 3: a[m][r] = sum_k x[m][k] * lora_a[r][k] -> ah[m][0:16] f16, pad 0.
// One block per token row; 16 lanes per rank entry x 16 K-slices; LDS reduce.
// ---------------------------------------------------------------------------
__global__ __launch_bounds__(256)
void lora_a_kernel(const float* __restrict__ x,
                   const float* __restrict__ la,
                   _Float16* __restrict__ ah) {
    __shared__ float red[256];
    const int m     = blockIdx.x;
    const int t     = threadIdx.x;
    const int r     = t & 15;
    const int slice = t >> 4;                          // 16 slices of 256
    const float* xp = x  + (size_t)m * IN_DIM + slice * 256;
    const float* ap = la + (size_t)r * IN_DIM + slice * 256;
    float acc = 0.f;
#pragma unroll 4
    for (int i = 0; i < 256; i += 4) {
        v4f xv = *(const v4f*)(xp + i);
        v4f av = *(const v4f*)(ap + i);
        acc += xv.x * av.x + xv.y * av.y + xv.z * av.z + xv.w * av.w;
    }
    red[t] = acc;
    __syncthreads();
    for (int s = 8; s >= 1; s >>= 1) {
        if (slice < s) red[t] += red[t + s * 16];
        __syncthreads();
    }
    if (t < 32)
        ah[(size_t)m * 32 + t] = (t < 16) ? (_Float16)red[t] : (_Float16)0.f;
}

// ---------------------------------------------------------------------------
// Kernel 4: main GEMM. out[m][n] = sum_k xh[m][k]*wh[n][k]  (+ LoRA K-step)
// Block: 256 thr = 8 waves, tile 128(M) x 128(N), K-step 32.
// Wave: 32x64 => 2x4 grid of 16x16 accumulators, v_wmma_f32_16x16x32_f16.
// A tile (needs fp32->f16 cvt) staged through VGPRs; B tile staged with
// GLOBAL_LOAD_ASYNC_TO_LDS_B128 (ASYNCcnt) -- no VGPR round-trip.
// Fragment layout per ISA: lane L holds row/col (L&15); K chunks
// [h*8, h*8+8) and [16+h*8, 16+h*8+8) with h = L>>4  => two ds_load_b128.
// ---------------------------------------------------------------------------
__global__ __launch_bounds__(256)
void qlora_wmma_gemm(const float* __restrict__ x,
                     const _Float16* __restrict__ wh,
                     const _Float16* __restrict__ ah,
                     const _Float16* __restrict__ bh,
                     float* __restrict__ out) {
    __shared__ _Float16 As[128 * LDA];
    __shared__ _Float16 Bs[128 * LDA];

    const int tid  = threadIdx.x;
    const int lane = tid & 31;
    const int wave = tid >> 5;
    const int wm   = (wave >> 1) * 32;                 // wave M offset in tile
    const int wn   = (wave & 1) * 64;                  // wave N offset in tile
    const int l15  = lane & 15;
    const int h    = lane >> 4;

    const int m0 = blockIdx.y * 128;
    const int n0 = blockIdx.x * 128;

    // staging assignment: 2 threads per row, 16 halves (32B of f16) each
    const int row2 = tid >> 1;                         // 0..127
    const int cg   = (tid & 1) * 16;                   // half-offset 0 / 16

    // LDS byte offsets for this thread's staging slots
    const unsigned asOff = lds_offset(&As[row2 * LDA + cg]);
    const unsigned bsOff = lds_offset(&Bs[row2 * LDA + cg]);

    v8f c[2][4];
    const v8f vzero = {};
#pragma unroll
    for (int i = 0; i < 2; ++i)
#pragma unroll
        for (int j = 0; j < 4; ++j) c[i][j] = vzero;

    const int KSTEPS = IN_DIM / 32;                    // 128 + 1 LoRA step

    for (int kt = 0; kt <= KSTEPS; ++kt) {
        __syncthreads();
        if (kt < KSTEPS) {
            const int kb = kt * 32;
            // ---- stage B: wh f16, async Global->LDS (2 x 16B per thread) ----
            const _Float16* wsrc = wh + (size_t)(n0 + row2) * IN_DIM + kb + cg;
            async_copy_b128(bsOff,      wsrc);
            async_copy_b128(bsOff + 16, wsrc + 8);
            // ---- stage A: x fp32 -> f16 via VGPRs ----
            const float* xs = x + (size_t)(m0 + row2) * IN_DIM + kb + cg;
            v4f f0 = *(const v4f*)(xs + 0);
            v4f f1 = *(const v4f*)(xs + 4);
            v4f f2 = *(const v4f*)(xs + 8);
            v4f f3 = *(const v4f*)(xs + 12);
            v8h ha = {(_Float16)f0.x, (_Float16)f0.y, (_Float16)f0.z, (_Float16)f0.w,
                      (_Float16)f1.x, (_Float16)f1.y, (_Float16)f1.z, (_Float16)f1.w};
            v8h hb = {(_Float16)f2.x, (_Float16)f2.y, (_Float16)f2.z, (_Float16)f2.w,
                      (_Float16)f3.x, (_Float16)f3.y, (_Float16)f3.z, (_Float16)f3.w};
            *(v8h*)&As[row2 * LDA + cg + 0] = ha;
            *(v8h*)&As[row2 * LDA + cg + 8] = hb;
            // prefetch next x K-tile (gfx1250 global_prefetch path)
            if (kt + 1 < KSTEPS) __builtin_prefetch(xs + 32, 0, 3);
        } else {
            // ---- LoRA K-step: both operands f16 passthrough -> async ----
            const _Float16* asrc = ah + (size_t)(m0 + row2) * 32 + cg;
            async_copy_b128(asOff,      asrc);
            async_copy_b128(asOff + 16, asrc + 8);
            const _Float16* bsrc = bh + (size_t)(n0 + row2) * 32 + cg;
            async_copy_b128(bsOff,      bsrc);
            async_copy_b128(bsOff + 16, bsrc + 8);
        }
        wait_asynccnt0();          // this wave's async LDS writes landed
        __syncthreads();           // publish all waves' tiles

        // ---- load fragments from LDS ----
        v16h a[2], b[4];
#pragma unroll
        for (int tm = 0; tm < 2; ++tm) {
            const int row = wm + tm * 16 + l15;
            v8h lo = *(const v8h*)&As[row * LDA + h * 8];
            v8h hi = *(const v8h*)&As[row * LDA + 16 + h * 8];
            a[tm] = __builtin_shufflevector(lo, hi, 0,1,2,3,4,5,6,7,8,9,10,11,12,13,14,15);
        }
#pragma unroll
        for (int tn = 0; tn < 4; ++tn) {
            const int col = wn + tn * 16 + l15;
            v8h lo = *(const v8h*)&Bs[col * LDA + h * 8];
            v8h hi = *(const v8h*)&Bs[col * LDA + 16 + h * 8];
            b[tn] = __builtin_shufflevector(lo, hi, 0,1,2,3,4,5,6,7,8,9,10,11,12,13,14,15);
        }
        // ---- 8 WMMAs ----
#pragma unroll
        for (int tm = 0; tm < 2; ++tm)
#pragma unroll
            for (int tn = 0; tn < 4; ++tn)
                c[tm][tn] = __builtin_amdgcn_wmma_f32_16x16x32_f16(
                    false, a[tm], false, b[tn], (short)0, c[tm][tn], false, false);
    }

    // ---- epilogue: C/D layout => m = base + h*8 + r, n = base + (lane&15) ----
#pragma unroll
    for (int tm = 0; tm < 2; ++tm)
#pragma unroll
        for (int tn = 0; tn < 4; ++tn) {
            const int n = n0 + wn + tn * 16 + l15;
            const int mb = m0 + wm + tm * 16 + h * 8;
#pragma unroll
            for (int r = 0; r < 8; ++r)
                out[(size_t)(mb + r) * OUT_DIM + n] = c[tm][tn][r];
        }
}

// ---------------------------------------------------------------------------
extern "C" void kernel_launch(void* const* d_in, const int* in_sizes, int n_in,
                              void* d_out, int out_size, void* d_ws, size_t ws_size,
                              hipStream_t stream) {
    const float* x      = (const float*)d_in[0];   // [4,2048,4096]
    const int*   wq     = (const int*)d_in[1];     // [4096,4096]
    const float* scales = (const float*)d_in[2];   // [4096,64]
    const float* la     = (const float*)d_in[3];   // [16,4096]
    const float* lb     = (const float*)d_in[4];   // [4096,16]
    float* out = (float*)d_out;                    // [4,2048,4096]

    _Float16* wh = (_Float16*)d_ws;                       // 32 MB
    _Float16* ah = wh + (size_t)OUT_DIM * IN_DIM;         // 512 KB
    _Float16* bh = ah + (size_t)M_TOK * 32;               // 256 KB

    dequant_nf4_kernel<<<(OUT_DIM * IN_DIM) / (256 * 8), 256, 0, stream>>>(wq, scales, wh);
    pack_lorab_kernel<<<OUT_DIM / 256, 256, 0, stream>>>(lb, bh);
    lora_a_kernel<<<M_TOK, 256, 0, stream>>>(x, la, ah);

    dim3 grid(OUT_DIM / 128, M_TOK / 128);                // (32, 64)
    qlora_wmma_gemm<<<grid, 256, 0, stream>>>(x, wh, ah, bh, out);
}
